// DGCNN_58317065945579
// MI455X (gfx1250) — matrix operand order repaired and checked by reading, not compiled, not run on hardware
//
#include <hip/hip_runtime.h>

// ---------------------------------------------------------------------------
// DGCNN forward for MI455X (gfx1250, wave32, WMMA + async LDS copies).
// All contractions use v_wmma_f32_16x16x32_f16 (f16 operands, f32 accum).
// ---------------------------------------------------------------------------

typedef __attribute__((ext_vector_type(16))) _Float16 v16h;
typedef __attribute__((ext_vector_type(8)))  _Float16 v8h;
typedef __attribute__((ext_vector_type(8)))  float    v8f;

#define BN_EPS 1e-3f
#define KNN 16
#define NEG_INF (-3.0e38f)

// ---------------------------------------------------------------------------
// WMMA fragment loaders. Contiguous k order used here for both A and B:
//  A (16x32, MxK): lane m=l&15, hi=l>>4; VGPR0..7 = K[k0+16*hi .. +15]
//  contiguous. -> two 16B contiguous loads.
//  B (32x16, KxN): lane n=l&15, hi=l>>4; VGPR0..7 = K[k0+16*hi .. +15]
//  contiguous. -> one 32B contiguous region.
// Requires k0 and row strides to be multiples of 8 halfs (16B aligned).
// ---------------------------------------------------------------------------
__device__ inline v16h load_a_frag(const _Float16* base, int rowStride, int k0, int lane) {
    const int m = lane & 15, hi = lane >> 4;
    const _Float16* p = base + (size_t)m * rowStride + k0 + 16 * hi;
    const v8h a0 = *reinterpret_cast<const v8h*>(p);
    const v8h a1 = *reinterpret_cast<const v8h*>(p + 8);
    v16h r;
#pragma unroll
    for (int i = 0; i < 8; ++i) { r[i] = a0[i]; r[i + 8] = a1[i]; }
    return r;
}

__device__ inline v16h load_b_frag(const _Float16* base, int rowStride, int k0, int lane) {
    const int n = lane & 15, hi = lane >> 4;
    const _Float16* p = base + (size_t)n * rowStride + k0 + 16 * hi;
    const v8h b0 = *reinterpret_cast<const v8h*>(p);
    const v8h b1 = *reinterpret_cast<const v8h*>(p + 8);
    v16h r;
#pragma unroll
    for (int i = 0; i < 8; ++i) { r[i] = b0[i]; r[i + 8] = b1[i]; }
    return r;
}

__device__ inline v8f zero8() {
    v8f z;
#pragma unroll
    for (int i = 0; i < 8; ++i) z[i] = 0.0f;
    return z;
}

// ---------------------------------------------------------------------------
// Async global->LDS 16B copy (CDNA5 GLOBAL_LOAD_ASYNC_TO_LDS_B128, GV mode).
// the LDS destination address comes from the VDST VGPR; the low 32 bits of a generic pointer to a
// __shared__ object are exactly the LDS byte offset. Tracked by ASYNCcnt.
// ---------------------------------------------------------------------------
__device__ inline void async_copy16(void* lds_ptr, const void* gptr) {
    uint32_t lds_off = (uint32_t)(uintptr_t)lds_ptr;
    uint64_t gaddr   = (uint64_t)(uintptr_t)gptr;
    asm volatile("global_load_async_to_lds_b128 %0, %1, off"
                 :: "v"(lds_off), "v"(gaddr)
                 : "memory");
}

__device__ inline void wait_async0() {
#if __has_builtin(__builtin_amdgcn_s_wait_asynccnt)
    __builtin_amdgcn_s_wait_asynccnt(0);
#else
    asm volatile("s_wait_asynccnt 0" ::: "memory");
#endif
}

// descending-sorted top-16 insertion
__device__ inline void tk_insert(float* v, int* id, float nv, int ni) {
    if (nv <= v[KNN - 1]) return;
    int p = KNN - 1;
    while (p > 0 && v[p - 1] < nv) { v[p] = v[p - 1]; id[p] = id[p - 1]; --p; }
    v[p] = nv; id[p] = ni;
}

// ---------------------------------------------------------------------------
// prep: f32 x -> f16, K padded 3->32 with zeros; per-point squared norm.
// ---------------------------------------------------------------------------
__global__ void prep_x_kernel(const float* __restrict__ x, _Float16* __restrict__ x0h,
                              float* __restrict__ sq, int BN) {
    int i = blockIdx.x * blockDim.x + threadIdx.x;
    if (i >= BN) return;
    float s = 0.0f;
#pragma unroll
    for (int c = 0; c < 32; ++c) {
        float val = (c < 3) ? x[i * 3 + c] : 0.0f;
        _Float16 h = (_Float16)val;
        x0h[(size_t)i * 32 + c] = h;
        float f = (float)h;
        s += f * f;   // padded lanes contribute 0
    }
    sq[i] = s;
}

// prep: W [Kdim, D] f32 -> Wt [D][Kpad] f16 (transposed, zero-padded K)
__global__ void prep_w_kernel(const float* __restrict__ W, _Float16* __restrict__ Wt,
                              int kdim, int kpad, int dout) {
    int i = blockIdx.x * blockDim.x + threadIdx.x;
    int tot = dout * kpad;
    if (i >= tot) return;
    int d = i / kpad, k = i - d * kpad;
    Wt[i] = (k < kdim) ? (_Float16)W[(size_t)k * dout + d] : (_Float16)0.0f;
}

// ---------------------------------------------------------------------------
// kNN: one workgroup handles 16 query rows of one batch. 8 waves sweep the
// 128 column tiles; each tile is a 16x16 Gram block via WMMA. Score =
// 2*dot - ||x_col||^2 (row norm dropped: constant per row). Per-lane top-16
// with LDS staging, then cross-wave merge. idx written per-batch (0..N-1).
// KPAD templated so the K loop is branch-free static WMMA clauses.
// ---------------------------------------------------------------------------
template <int KPAD>
__global__ __launch_bounds__(256)
void knn_kernel(const _Float16* __restrict__ src, int rowStride, int chanOff,
                const float* __restrict__ sq,
                int* __restrict__ idxOut, int nPerB) {
    constexpr int NCH = KPAD / 32;
    __shared__ __align__(16) float tileBuf[8][16][17];
    __shared__ float pv[8 * 32 * KNN];
    __shared__ int   pi[8 * 32 * KNN];

    const int tid = threadIdx.x, lane = tid & 31, wid = tid >> 5;
    const int rowTilesPerB = nPerB / 16;
    const int b  = blockIdx.x / rowTilesPerB;
    const int rt = blockIdx.x % rowTilesPerB;
    const int rowBase = b * nPerB + rt * 16;

    // A fragments for the 16 query rows (shared by all tiles of this wave)
    v16h af[NCH];
#pragma unroll
    for (int c = 0; c < NCH; ++c)
        af[c] = load_a_frag(src + (size_t)rowBase * rowStride + chanOff,
                            rowStride, c * 32, lane);

    float tv[KNN]; int ti[KNN];
#pragma unroll
    for (int j = 0; j < KNN; ++j) { tv[j] = NEG_INF; ti[j] = 0; }

    const int selRow  = lane & 15;
    const int selHalf = lane >> 4;
    const int nTiles  = nPerB / 16;          // 128: uniform trip count per wave

    for (int t = wid; t < nTiles; t += 8) {
        // prefetch this wave's next column tile while we do the WMMAs
        if (t + 8 < nTiles)
            __builtin_prefetch(src + (size_t)(b * nPerB + (t + 8) * 16) * rowStride + chanOff, 0, 1);
        v8f acc = zero8();
#pragma unroll
        for (int c = 0; c < NCH; ++c) {
            v16h bf = load_b_frag(src + (size_t)(b * nPerB + t * 16) * rowStride + chanOff,
                                  rowStride, c * 32, lane);
            acc = __builtin_amdgcn_wmma_f32_16x16x32_f16(
                      false, af[c], false, bf, (short)0, acc, false, false);
        }
        const int col = t * 16 + (lane & 15);
        const float sqc = sq[b * nPerB + col];
#pragma unroll
        for (int v = 0; v < 8; ++v)
            tileBuf[wid][v + 8 * (lane >> 4)][lane & 15] = 2.0f * acc[v] - sqc;
        __syncthreads();
#pragma unroll
        for (int j = 0; j < 8; ++j) {
            int c2 = selHalf * 8 + j;
            tk_insert(tv, ti, tileBuf[wid][selRow][c2], t * 16 + c2);
        }
        __syncthreads();
    }

    const int pbase = (wid * 32 + lane) * KNN;
#pragma unroll
    for (int j = 0; j < KNN; ++j) { pv[pbase + j] = tv[j]; pi[pbase + j] = ti[j]; }
    __syncthreads();

    if (tid < 16) {
        float mv[KNN]; int mi[KNN];
#pragma unroll
        for (int j = 0; j < KNN; ++j) { mv[j] = NEG_INF; mi[j] = 0; }
        for (int w = 0; w < 8; ++w)
            for (int h = 0; h < 2; ++h) {
                int base2 = (w * 32 + h * 16 + tid) * KNN;
                for (int j = 0; j < KNN; ++j) {
                    float val = pv[base2 + j];
                    if (val <= mv[KNN - 1]) break;  // lists are sorted desc
                    tk_insert(mv, mi, val, pi[base2 + j]);
                }
            }
#pragma unroll
        for (int j = 0; j < KNN; ++j)
            idxOut[(size_t)(rowBase + tid) * KNN + j] = mi[j];
    }
}

// ---------------------------------------------------------------------------
// Edge conv: one workgroup per point. Gather 16x(2C) edge features into LDS
// (K zero-padded to KPAD, row stride padded for bank spread). For KPAD>=64
// (cin = KPAD/2, 16B-aligned rows) the gather is issued as per-lane
// GLOBAL_LOAD_ASYNC_TO_LDS_B128 copies drained with s_wait_asynccnt. Each
// wave then produces 16-column output tiles via WMMA against pre-transposed
// W, applies BN+ReLU, and max-reduces over the 16 neighbors (register max
// over the 8 accumulator rows + shfl_xor(16) across wave halves). Also
// accumulates the point's squared norm for the next layer's kNN via a shared
// float atomic (ds_add_f32).
// ---------------------------------------------------------------------------
template <int KPAD>
__global__ __launch_bounds__(256)
void edge_kernel(const _Float16* __restrict__ src, int rowStride, int chanOff, int cin,
                 const _Float16* __restrict__ wt, int dout,
                 const float* __restrict__ bnG, const float* __restrict__ bnB,
                 const float* __restrict__ bnM, const float* __restrict__ bnV,
                 const int* __restrict__ idx,
                 _Float16* __restrict__ outH, int outOff,
                 float* __restrict__ sqOut, int nPerB) {
    constexpr int ESTRIDE = KPAD + 8;      // halfs; multiple of 8 -> 16B rows
    __shared__ __align__(16) _Float16 eA[16 * ESTRIDE];
    __shared__ int   sIdx[16];
    __shared__ float sSq;

    const int tid = threadIdx.x, lane = tid & 31, wid = tid >> 5;
    const int pt = blockIdx.x;                 // flat b*N + n
    const int b  = pt / nPerB;

    if (tid < 16) sIdx[tid] = idx[(size_t)pt * KNN + tid];
    if (tid == 0) sSq = 0.0f;
    __syncthreads();

    if constexpr (KPAD >= 64) {
        // cin == KPAD/2, rows are whole 16B chunks: async DMA into LDS.
        constexpr int CPR = KPAD / 8;          // 16B chunks per edge row
        constexpr int TOT = 16 * CPR;
#pragma unroll
        for (int e0 = 0; e0 < TOT; e0 += 256) {
            int e = e0 + tid;
            if (e < TOT) {
                int j = e / CPR, cc = e - j * CPR;
                int c = cc * 8;
                const _Float16* gp =
                    (c < cin) ? src + (size_t)pt * rowStride + chanOff + c
                              : src + (size_t)(b * nPerB + sIdx[j]) * rowStride + chanOff + (c - cin);
                async_copy16(&eA[j * ESTRIDE + c], gp);
            }
        }
        wait_async0();
    } else {
        // ragged layer-1 path (cin=3, zero-padded to KPAD)
        const int tot = 16 * KPAD;
        for (int e = tid; e < tot; e += 256) {
            int j = e / KPAD, c = e - j * KPAD;
            _Float16 val = (_Float16)0.0f;
            if (c < cin)
                val = src[(size_t)pt * rowStride + chanOff + c];
            else if (c < 2 * cin)
                val = src[(size_t)(b * nPerB + sIdx[j]) * rowStride + chanOff + (c - cin)];
            eA[j * ESTRIDE + c] = val;
        }
    }
    __syncthreads();

    const int nTiles = dout >> 4;
    for (int t = wid; t < nTiles; t += 8) {
        v8f acc = zero8();
#pragma unroll
        for (int k0 = 0; k0 < KPAD; k0 += 32) {
            v16h a  = load_a_frag(eA, ESTRIDE, k0, lane);
            v16h bf = load_b_frag(wt + (size_t)(t * 16) * KPAD, KPAD, k0, lane);
            acc = __builtin_amdgcn_wmma_f32_16x16x32_f16(
                      false, a, false, bf, (short)0, acc, false, false);
        }
        const int col = t * 16 + (lane & 15);
        const float sc = bnG[col] * rsqrtf(bnV[col] + BN_EPS);
        const float mc = bnM[col], bc = bnB[col];
        float rmax = NEG_INF;
#pragma unroll
        for (int v = 0; v < 8; ++v) {
            float y = fmaxf((acc[v] - mc) * sc + bc, 0.0f);
            rmax = fmaxf(rmax, y);
        }
        rmax = fmaxf(rmax, __shfl_xor(rmax, 16, 32));  // combine rows 0-7 / 8-15
        if (lane < 16) {
            outH[(size_t)pt * 512 + outOff + col] = (_Float16)rmax;
            atomicAdd(&sSq, rmax * rmax);
        }
    }
    __syncthreads();
    if (tid == 0) sqOut[pt] = sSq;
}

// ---------------------------------------------------------------------------
// Final: out = relu(bn(concat(x1..x4) @ W5)). M=B*N=16384, K=512.
// Workgroup = 16 rows x 128 cols (8 waves x one 16x16 tile each).
// ---------------------------------------------------------------------------
__global__ __launch_bounds__(256)
void final_kernel(const _Float16* __restrict__ xall, const _Float16* __restrict__ w5t,
                  const float* __restrict__ bnG, const float* __restrict__ bnB,
                  const float* __restrict__ bnM, const float* __restrict__ bnV,
                  float* __restrict__ out) {
    const int lane = threadIdx.x & 31, wid = threadIdx.x >> 5;
    const int rowBase = blockIdx.x * 16;
    const int colBase = blockIdx.y * 128 + wid * 16;

    v8f acc = zero8();
#pragma unroll 4
    for (int k0 = 0; k0 < 512; k0 += 32) {
        if (k0 + 32 < 512) {
            __builtin_prefetch(xall + (size_t)rowBase * 512 + k0 + 32, 0, 1);
            __builtin_prefetch(w5t + (size_t)colBase * 512 + k0 + 32, 0, 1);
        }
        v16h a  = load_a_frag(xall + (size_t)rowBase * 512, 512, k0, lane);
        v16h bf = load_b_frag(w5t + (size_t)colBase * 512, 512, k0, lane);
        acc = __builtin_amdgcn_wmma_f32_16x16x32_f16(
                  false, a, false, bf, (short)0, acc, false, false);
    }
    const int col = colBase + (lane & 15);
    const int hi  = lane >> 4;
    const float sc = bnG[col] * rsqrtf(bnV[col] + BN_EPS);
    const float mc = bnM[col], bc = bnB[col];
#pragma unroll
    for (int v = 0; v < 8; ++v) {
        int row = rowBase + v + 8 * hi;
        out[(size_t)row * 512 + col] = fmaxf((acc[v] - mc) * sc + bc, 0.0f);
    }
}

// ---------------------------------------------------------------------------
extern "C" void kernel_launch(void* const* d_in, const int* in_sizes, int n_in,
                              void* d_out, int out_size, void* d_ws, size_t ws_size,
                              hipStream_t stream) {
    (void)n_in; (void)out_size; (void)ws_size;

    const float* x = (const float*)d_in[0];
    const float *W[5], *G[5], *Bb[5], *Mm[5], *Vv[5];
    for (int j = 0; j < 5; ++j) {
        int base = 1 + j * 5;
        W[j]  = (const float*)d_in[base + 0];
        G[j]  = (const float*)d_in[base + 1];
        Bb[j] = (const float*)d_in[base + 2];
        Mm[j] = (const float*)d_in[base + 3];
        Vv[j] = (const float*)d_in[base + 4];
    }

    const int nPerB = 2048;
    const int BN = in_sizes[0] / 3;          // B * N = 16384

    // ---- workspace carve (all chunks 16B-aligned sizes) ----
    uint8_t* ws = (uint8_t*)d_ws;
    size_t o = 0;
    _Float16* xall = (_Float16*)(ws + o); o += (size_t)BN * 512 * 2; // 16 MB
    _Float16* x0h  = (_Float16*)(ws + o); o += (size_t)BN * 32 * 2;  // 1 MB
    float*    sq   = (float*)(ws + o);    o += (size_t)BN * 4;       // 64 KB
    int*      idx  = (int*)(ws + o);      o += (size_t)BN * KNN * 4; // 1 MB
    const int kdims[5] = {6, 128, 128, 256, 512};
    const int kpads[5] = {32, 128, 128, 256, 512};
    const int douts[5] = {64, 64, 128, 256, 512};
    _Float16* wt[5];
    for (int j = 0; j < 5; ++j) { wt[j] = (_Float16*)(ws + o); o += (size_t)kpads[j] * douts[j] * 2; }

    // ---- preparation ----
    prep_x_kernel<<<(BN + 255) / 256, 256, 0, stream>>>(x, x0h, sq, BN);
    for (int j = 0; j < 5; ++j) {
        int tot = douts[j] * kpads[j];
        prep_w_kernel<<<(tot + 255) / 256, 256, 0, stream>>>(W[j], wt[j], kdims[j], kpads[j], douts[j]);
    }

    const int knnBlocks = BN / 16;

    // ---- block 1: x(3) -> x1(64) at channel offset 0 ----
    knn_kernel<32><<<knnBlocks, 256, 0, stream>>>(x0h, 32, 0, sq, idx, nPerB);
    edge_kernel<32><<<BN, 256, 0, stream>>>(x0h, 32, 0, 3, wt[0], 64,
                                            G[0], Bb[0], Mm[0], Vv[0], idx, xall, 0, sq, nPerB);
    // ---- block 2: x1(64)@0 -> x2(64)@64 ----
    knn_kernel<64><<<knnBlocks, 256, 0, stream>>>(xall, 512, 0, sq, idx, nPerB);
    edge_kernel<128><<<BN, 256, 0, stream>>>(xall, 512, 0, 64, wt[1], 64,
                                             G[1], Bb[1], Mm[1], Vv[1], idx, xall, 64, sq, nPerB);
    // ---- block 3: x2(64)@64 -> x3(128)@128 ----
    knn_kernel<64><<<knnBlocks, 256, 0, stream>>>(xall, 512, 64, sq, idx, nPerB);
    edge_kernel<128><<<BN, 256, 0, stream>>>(xall, 512, 64, 64, wt[2], 128,
                                             G[2], Bb[2], Mm[2], Vv[2], idx, xall, 128, sq, nPerB);
    // ---- block 4: x3(128)@128 -> x4(256)@256 ----
    knn_kernel<128><<<knnBlocks, 256, 0, stream>>>(xall, 512, 128, sq, idx, nPerB);
    edge_kernel<256><<<BN, 256, 0, stream>>>(xall, 512, 128, 128, wt[3], 256,
                                             G[3], Bb[3], Mm[3], Vv[3], idx, xall, 256, sq, nPerB);

    // ---- final 512x512 GEMM + BN + ReLU ----
    dim3 fg(BN / 16, 4);
    final_kernel<<<fg, 256, 0, stream>>>(xall, wt[4], G[4], Bb[4], Mm[4], Vv[4], (float*)d_out);
}
